// GNN_75754633167642
// MI455X (gfx1250) — matrix operand order repaired and checked by reading, not compiled, not run on hardware
//
#include <hip/hip_runtime.h>

// GCN: N=100000 nodes, E=3.2M edges, dims 768 -> 16 -> 16 -> 16 -> 3.
// Memory-bound workload: WMMA f16 (f32 acc) for the dense per-node GEMMs
// with B staged in LDS in fragment layout; f32 global atomics for the
// edge scatter-add.

typedef __attribute__((ext_vector_type(16))) _Float16 v16h;
typedef __attribute__((ext_vector_type(8)))  float    v8f;

#define HIDF 16   // hidden width / feature stride of intermediates

// ---------- degree / norm ----------

__global__ void init_deg_kernel(float* __restrict__ deg, int n) {
  int i = blockIdx.x * blockDim.x + threadIdx.x;
  if (i < n) deg[i] = 1.0f;  // self-loop contributes 1 to every node's degree
}

__global__ void deg_accum_kernel(const long long* __restrict__ ei,
                                 float* __restrict__ deg, int E) {
  int e = blockIdx.x * blockDim.x + threadIdx.x;
  if (e < E) {
    int d = (int)ei[(long long)E + e];  // dst row of edge_index
    atomicAdd(&deg[d], 1.0f);
  }
}

__global__ void rsqrt_kernel(float* __restrict__ dinv, int n) {
  int i = blockIdx.x * blockDim.x + threadIdx.x;
  if (i < n) dinv[i] = rsqrtf(dinv[i]);  // deg >= 1 always (self-loop)
}

__global__ void zero_kernel(float* __restrict__ p, int n) {
  int i = blockIdx.x * blockDim.x + threadIdx.x;
  if (i < n) p[i] = 0.0f;
}

// ---------- WMMA GEMM: T[nrows x 16] = X[nrows x K] @ W[K x NC] ----------
// K, NC compile-time. One wave per 16-row tile, 8 waves/block.
// B is converted to f16 once per block into LDS, pre-swizzled into the exact
// per-lane fragment layout: ldsB[chunk][lane][16 halves] (32 B/lane/chunk).
// NC <= 16; output columns >= NC come out 0 (W zero-padded during staging).

template <int K, int NC>
__global__ __launch_bounds__(256)
void gemm_wmma_kernel(const float* __restrict__ X, const float* __restrict__ W,
                      float* __restrict__ T, int nrows) {
  constexpr int NCHUNK = (K + 31) / 32;
  __shared__ _Float16 ldsB[NCHUNK * 32 * 16];

  const int lane = threadIdx.x & 31;
  const int wave = threadIdx.x >> 5;
  const int tile = blockIdx.x * (blockDim.x >> 5) + wave;
  const int ntiles = (nrows + 15) >> 4;

  // --- stage B into LDS in fragment layout (all threads cooperate) ---
  // slot = (chunk, lane): lane l holds col l&15, K kb..kb+7 and kb+16..kb+23
  // where kb = chunk*32 + (l>=16)*8. Zero-pad K>=K and col>=NC branchlessly.
  for (int slot = threadIdx.x; slot < NCHUNK * 32; slot += blockDim.x) {
    const int kc  = slot >> 5;
    const int l   = slot & 31;
    const int col = l & 15;
    const int kb  = kc * 32 + ((l >> 4) << 3);
    const int colc = (col < NC) ? col : NC - 1;  // clamped in-bounds index
    #pragma unroll
    for (int i = 0; i < 16; ++i) {
      const int kk  = kb + i + ((i >> 3) << 3);
      const int kkc = (kk < K) ? kk : K - 1;
      const float v = W[kkc * NC + colc];
      const bool ok = (kk < K) && (col < NC);
      ldsB[slot * 16 + i] = ok ? (_Float16)v : (_Float16)0.0f;
    }
  }
  __syncthreads();

  if (tile >= ntiles) return;  // uniform per-wave exit: EXEC stays all-ones

  // A fragment: lane holds row (lane&15); halves 0..7 = K khalf..khalf+7,
  // halves 8..15 = K 16+khalf..16+khalf+7, khalf = (lane>=16)*8.
  int rowA = tile * 16 + (lane & 15);
  if (rowA >= nrows) rowA = nrows - 1;  // clamp (stores are guarded anyway)
  const int khalfA = (lane >> 4) * 8;
  const float* __restrict__ xrow = X + (long long)rowA * K;
  const _Float16* __restrict__ bfrag = ldsB + lane * 16;

  v8f c = {};
  #pragma unroll 4
  for (int kc = 0; kc < NCHUNK; ++kc) {
    const int k = kc * 32;
    v16h a;
    if constexpr (K % 32 == 0) {
      const float4 x0 = *(const float4*)(xrow + k + khalfA);
      const float4 x1 = *(const float4*)(xrow + k + khalfA + 4);
      const float4 x2 = *(const float4*)(xrow + k + 16 + khalfA);
      const float4 x3 = *(const float4*)(xrow + k + 16 + khalfA + 4);
      __builtin_prefetch(xrow + k + 32, 0, 1);  // speculative: safe at tail
      a[0]  = (_Float16)x0.x; a[1]  = (_Float16)x0.y;
      a[2]  = (_Float16)x0.z; a[3]  = (_Float16)x0.w;
      a[4]  = (_Float16)x1.x; a[5]  = (_Float16)x1.y;
      a[6]  = (_Float16)x1.z; a[7]  = (_Float16)x1.w;
      a[8]  = (_Float16)x2.x; a[9]  = (_Float16)x2.y;
      a[10] = (_Float16)x2.z; a[11] = (_Float16)x2.w;
      a[12] = (_Float16)x3.x; a[13] = (_Float16)x3.y;
      a[14] = (_Float16)x3.z; a[15] = (_Float16)x3.w;
    } else {
      // K == 16: halves 0..7 = K khalfA..khalfA+7; halves 8..15 (K 16..31) = 0
      static_assert(K == 16, "partial-K path assumes K==16");
      const float4 x0 = *(const float4*)(xrow + khalfA);
      const float4 x1 = *(const float4*)(xrow + khalfA + 4);
      a[0] = (_Float16)x0.x; a[1] = (_Float16)x0.y;
      a[2] = (_Float16)x0.z; a[3] = (_Float16)x0.w;
      a[4] = (_Float16)x1.x; a[5] = (_Float16)x1.y;
      a[6] = (_Float16)x1.z; a[7] = (_Float16)x1.w;
      #pragma unroll
      for (int i = 8; i < 16; ++i) a[i] = (_Float16)0.0f;
    }
    const v16h b = *(const v16h*)(bfrag + kc * 32 * 16);
    // (neg_a, A, neg_b, B, c_mod, C, reuse_a, reuse_b)
    c = __builtin_amdgcn_wmma_f32_16x16x32_f16(false, a, false, b,
                                               (short)0, c, false, false);
  }

  // C/D layout: VGPR j -> row j + (lane>=16)*8, col lane&15
  const int colO  = lane & 15;
  const int rbase = tile * 16 + (lane >> 4) * 8;
  #pragma unroll
  for (int j = 0; j < 8; ++j) {
    const int r = rbase + j;
    if (r < nrows) T[(long long)r * HIDF + colO] = c[j];
  }
}

// ---------- edge scatter-add: A[dst] += dinv[src]*dinv[dst] * T[src] ----------

__global__ void aggregate_kernel(const long long* __restrict__ ei,
                                 const float* __restrict__ dinv,
                                 const float* __restrict__ T,
                                 float* __restrict__ A, int E) {
  int e = blockIdx.x * blockDim.x + threadIdx.x;
  if (e >= E) return;
  const int s = (int)ei[e];
  const int d = (int)ei[(long long)E + e];
  const float nm = dinv[s] * dinv[d];
  const float4* __restrict__ ts = (const float4*)(T + (long long)s * HIDF);
  float* __restrict__ ad = A + (long long)d * HIDF;
  #pragma unroll
  for (int q = 0; q < 4; ++q) {
    const float4 v = ts[q];
    atomicAdd(ad + 4 * q + 0, nm * v.x);
    atomicAdd(ad + 4 * q + 1, nm * v.y);
    atomicAdd(ad + 4 * q + 2, nm * v.z);
    atomicAdd(ad + 4 * q + 3, nm * v.w);
  }
}

// ---------- epilogue: fold self-loop term, bias, ReLU ----------

__global__ void bias_relu_kernel(const float* __restrict__ A,
                                 const float* __restrict__ T,
                                 const float* __restrict__ dinv,
                                 const float* __restrict__ b,
                                 float* __restrict__ H, int n) {
  int idx = blockIdx.x * blockDim.x + threadIdx.x;
  if (idx >= n * HIDF) return;
  const int i = idx >> 4;
  const int f = idx & 15;
  const float di = dinv[i];
  const float v = A[idx] + di * di * T[idx] + b[f];
  H[idx] = v > 0.0f ? v : 0.0f;
}

// ---------- final: self-loop + bias + log_softmax over 3 classes ----------

__global__ void logsoftmax_kernel(const float* __restrict__ A,
                                  const float* __restrict__ T,
                                  const float* __restrict__ dinv,
                                  const float* __restrict__ b,
                                  float* __restrict__ out, int n) {
  int i = blockIdx.x * blockDim.x + threadIdx.x;
  if (i >= n) return;
  const float di2 = dinv[i] * dinv[i];
  const long long base = (long long)i * HIDF;
  const float v0 = A[base + 0] + di2 * T[base + 0] + b[0];
  const float v1 = A[base + 1] + di2 * T[base + 1] + b[1];
  const float v2 = A[base + 2] + di2 * T[base + 2] + b[2];
  const float m = fmaxf(v0, fmaxf(v1, v2));
  const float lse = m + logf(expf(v0 - m) + expf(v1 - m) + expf(v2 - m));
  out[(long long)i * 3 + 0] = v0 - lse;
  out[(long long)i * 3 + 1] = v1 - lse;
  out[(long long)i * 3 + 2] = v2 - lse;
}

// ---------- launch ----------

extern "C" void kernel_launch(void* const* d_in, const int* in_sizes, int n_in,
                              void* d_out, int out_size, void* d_ws, size_t ws_size,
                              hipStream_t stream) {
  (void)n_in; (void)out_size; (void)ws_size;
  const float*     x  = (const float*)d_in[0];
  const long long* ei = (const long long*)d_in[1];  // int64 edge_index [2,E]
  const float* W1 = (const float*)d_in[2]; const float* b1 = (const float*)d_in[3];
  const float* W2 = (const float*)d_in[4]; const float* b2 = (const float*)d_in[5];
  const float* W3 = (const float*)d_in[6]; const float* b3 = (const float*)d_in[7];
  const float* W4 = (const float*)d_in[8]; const float* b4 = (const float*)d_in[9];
  float* out = (float*)d_out;

  const int IN_DIM = 768;
  const int N = in_sizes[0] / IN_DIM;  // 100000
  const int E = in_sizes[1] / 2;       // 3200000

  // workspace layout (floats): dinv[N] | T[N*16] | A[N*16] | H[N*16]
  float* dinv = (float*)d_ws;
  float* T = dinv + N;
  float* A = T + (size_t)N * HIDF;
  float* H = A + (size_t)N * HIDF;

  const int TPB = 256;
  const int gb_n  = (N + TPB - 1) / TPB;
  const int gb_e  = (E + TPB - 1) / TPB;
  const int gb_nf = (N * HIDF + TPB - 1) / TPB;
  const int ntiles = (N + 15) / 16;
  const int gb_gemm = (ntiles + 7) / 8;  // 8 waves (one tile each) per block

  // GCN normalization: deg = 1 + in-degree; dinv = deg^-1/2
  init_deg_kernel<<<gb_n, TPB, 0, stream>>>(dinv, N);
  deg_accum_kernel<<<gb_e, TPB, 0, stream>>>(ei, dinv, E);
  rsqrt_kernel<<<gb_n, TPB, 0, stream>>>(dinv, N);

  // layer 1: 768 -> 16
  gemm_wmma_kernel<768, 16><<<gb_gemm, TPB, 0, stream>>>(x, W1, T, N);
  zero_kernel<<<gb_nf, TPB, 0, stream>>>(A, N * HIDF);
  aggregate_kernel<<<gb_e, TPB, 0, stream>>>(ei, dinv, T, A, E);
  bias_relu_kernel<<<gb_nf, TPB, 0, stream>>>(A, T, dinv, b1, H, N);

  // layer 2: 16 -> 16
  gemm_wmma_kernel<16, 16><<<gb_gemm, TPB, 0, stream>>>(H, W2, T, N);
  zero_kernel<<<gb_nf, TPB, 0, stream>>>(A, N * HIDF);
  aggregate_kernel<<<gb_e, TPB, 0, stream>>>(ei, dinv, T, A, E);
  bias_relu_kernel<<<gb_nf, TPB, 0, stream>>>(A, T, dinv, b2, H, N);

  // layer 3: 16 -> 16
  gemm_wmma_kernel<16, 16><<<gb_gemm, TPB, 0, stream>>>(H, W3, T, N);
  zero_kernel<<<gb_nf, TPB, 0, stream>>>(A, N * HIDF);
  aggregate_kernel<<<gb_e, TPB, 0, stream>>>(ei, dinv, T, A, E);
  bias_relu_kernel<<<gb_nf, TPB, 0, stream>>>(A, T, dinv, b3, H, N);

  // layer 4: 16 -> 3 (W4 zero-padded to 16 cols during staging)
  gemm_wmma_kernel<16, 3><<<gb_gemm, TPB, 0, stream>>>(H, W4, T, N);
  zero_kernel<<<gb_nf, TPB, 0, stream>>>(A, N * HIDF);
  aggregate_kernel<<<gb_e, TPB, 0, stream>>>(ei, dinv, T, A, E);
  logsoftmax_kernel<<<gb_n, TPB, 0, stream>>>(A, T, dinv, b4, out, N);
}
